// FastRCNNRelationFCHead_32959579029985
// MI455X (gfx1250) — matrix operand order repaired and checked by reading, not compiled, not run on hardware
//
#include <hip/hip_runtime.h>
#include <hip/hip_bf16.h>
#include <stdint.h>

// MI455X / gfx1250 implementation of the FastRCNN Relation FC head.
//
// - Output projection hoisted through attention: (A_g@h)@Wo_g^T == A_g@(h@Wo_g^T)
//   => attention apply shrinks from 16x(1024^3) to 2x 16x(1024x1024x64) GEMMs.
// - All dense math on v_wmma_f32_16x16x32_f16 (f32 accumulate), wave32.
// - Global->LDS staging uses GLOBAL_LOAD_ASYNC_TO_LDS_B128 (ASYNCcnt path)
//   when the toolchain declares the builtin; otherwise falls back to
//   load+ds_store with global_prefetch_b8 hints.

typedef __attribute__((ext_vector_type(16))) _Float16 v16h;
typedef __attribute__((ext_vector_type(8)))  float    v8f;
typedef __attribute__((ext_vector_type(4)))  _Float16 v4h;

// Builtin-compatible pointer types for the async global->LDS copy:
// param0 is 'int __vector(4) __device__ *' (addrspace(1)), param1 the LDS dest.
typedef int v4i_b128 __attribute__((vector_size(16)));
typedef __attribute__((address_space(1))) v4i_b128 gv4i;
typedef __attribute__((address_space(3))) v4i_b128 sv4i;

#define R_DIM   1024
#define G_DIM   16
#define D_DIM   1024
#define E_DIM   64
#define FLAT_DIM (256 * 7 * 7)

#if defined(__HIP_DEVICE_COMPILE__) && __has_builtin(__builtin_amdgcn_global_load_async_to_lds_b128)
#define USE_ASYNC_LDS 1
#endif

// Stage 16 bytes global -> LDS.
__device__ __forceinline__ void stage16(_Float16* lds, const _Float16* g) {
#if defined(USE_ASYNC_LDS)
    __builtin_amdgcn_global_load_async_to_lds_b128(
        (gv4i*)(uintptr_t)g,
        (sv4i*)(unsigned)(uintptr_t)lds,
        0, 0);
#else
    *(uint4*)lds = *(const uint4*)g;
#endif
}

__device__ __forceinline__ void stage_wait() {
#if defined(USE_ASYNC_LDS)
#if __has_builtin(__builtin_amdgcn_s_wait_asynccnt)
    __builtin_amdgcn_s_wait_asynccnt(0);
#else
    asm volatile("s_wait_asynccnt 0" ::: "memory");
#endif
#endif
}

// ---------------------------------------------------------------------------
// f32 -> f16 conversion (x4 vectorized; all tensor sizes are multiples of 4)
// ---------------------------------------------------------------------------
__global__ void __launch_bounds__(256)
cvt_f32_to_f16(const float* __restrict__ src, _Float16* __restrict__ dst, int n4) {
    int i = blockIdx.x * blockDim.x + threadIdx.x;
    if (i >= n4) return;
    float4 v = ((const float4*)src)[i];
    v4h h;
    h.x = (_Float16)v.x; h.y = (_Float16)v.y;
    h.z = (_Float16)v.z; h.w = (_Float16)v.w;
    ((v4h*)dst)[i] = h;
}

// ---------------------------------------------------------------------------
// WMMA GEMM:  C[m,n] = act( alpha * sum_k A[m,k]*B[n,k] + bias[n] )
// A: [M,K] f16 row-major (lda); B: [N,K] f16 row-major (ldb)  => C = A@B^T.
// Block tile 128 x BN, K-slab 64 (2 WMMA sub-steps per barrier).
// 8 waves (wave32): wave tile 32 x (BN/2) => 2 x (BN/32) accumulators.
// blockIdx.z batches with element strides (grouped attention GEMMs).
// ---------------------------------------------------------------------------
#define BM 128
#define BK 64
#define LDSS (BK + 8)   // padded row stride (72 halves = 144B) to spread banks

template<int BN>
__global__ void __launch_bounds__(256)
gemm_f16_wmma(const _Float16* __restrict__ A, const _Float16* __restrict__ B,
              const float* __restrict__ bias,
              float* __restrict__ C32, _Float16* __restrict__ C16,
              int M, int N, int K, int lda, int ldb, int ldc,
              float alpha, int act,
              long long batchA, long long batchB, long long batchC,
              long long batchBias)
{
    constexpr int WN = BN / 2;    // wave N extent (64 or 32)
    constexpr int NT = WN / 16;   // WMMA n-tiles per wave (4 or 2)

    __shared__ _Float16 sA[BM * LDSS];
    __shared__ _Float16 sB[BN * LDSS];

    const int z = blockIdx.z;
    A += (long long)z * batchA;
    B += (long long)z * batchB;
    if (C32) C32 += (long long)z * batchC;
    if (C16) C16 += (long long)z * batchC;
    if (bias) bias += (long long)z * batchBias;

    const int tid  = threadIdx.x;
    const int lane = tid & 31;
    const int wid  = tid >> 5;
    const int wm   = wid >> 1;       // 0..3
    const int wn   = wid & 1;        // 0..1
    const int r16  = lane & 15;
    const int hi   = lane >> 4;

    const int m0 = blockIdx.y * BM;
    const int n0 = blockIdx.x * BN;

    // staging maps (256 threads, 16B chunks)
    const int ar  = tid >> 1;            // A row 0..127
    const int ac0 = (tid & 1) * 32;      // A col base {0,32}; 4 chunks
    constexpr int TPR_B = 256 / BN;      // threads per B row (2 or 4)
    constexpr int BCH   = 8 / TPR_B;     // 16B chunks per thread for B (4 or 2)
    const int br  = tid / TPR_B;
    const int bc0 = (tid % TPR_B) * (BCH * 8);

    v8f acc[2][NT] = {};
    union F16Frag { v16h v; uint4 u[2]; };

    const int kst = K / BK;
    for (int ks = 0; ks < kst; ++ks) {
        const int k0 = ks * BK;
        const _Float16* ga = A + (long long)(m0 + ar) * lda + (k0 + ac0);
        _Float16*       la = &sA[ar * LDSS + ac0];
#pragma unroll
        for (int j = 0; j < 4; ++j) stage16(la + j * 8, ga + j * 8);

        const _Float16* gb = B + (long long)(n0 + br) * ldb + (k0 + bc0);
        _Float16*       lb = &sB[br * LDSS + bc0];
#pragma unroll
        for (int j = 0; j < BCH; ++j) stage16(lb + j * 8, gb + j * 8);

#if !defined(USE_ASYNC_LDS)
        if (ks + 1 < kst) {   // hint next K slab (global_prefetch_b8)
            __builtin_prefetch(ga + BK, 0, 1);
            __builtin_prefetch(gb + BK, 0, 1);
        }
#endif
        stage_wait();         // this wave's async copies complete
        __syncthreads();      // all waves' staged data visible

#pragma unroll
        for (int kk = 0; kk < BK; kk += 32) {
            F16Frag af[2], bf[NT];
            // A frag (16x32): hi=0 lanes hold K 0-7/16-23, hi=1 hold 8-15/24-31
#pragma unroll
            for (int tm = 0; tm < 2; ++tm) {
                const _Float16* pa = &sA[(wm * 32 + tm * 16 + r16) * LDSS + kk];
                af[tm].u[0] = *(const uint4*)(pa + hi * 8);
                af[tm].u[1] = *(const uint4*)(pa + 16 + hi * 8);
            }
            // B frag (32x16): hi=0 lanes hold K 0-7/16-23, hi=1 hold 8-15/24-31 of column r16
#pragma unroll
            for (int tn = 0; tn < NT; ++tn) {
                const _Float16* pb = &sB[(wn * WN + tn * 16 + r16) * LDSS + kk];
                bf[tn].u[0] = *(const uint4*)(pb + hi * 8);
                bf[tn].u[1] = *(const uint4*)(pb + 16 + hi * 8);
            }
#pragma unroll
            for (int tm = 0; tm < 2; ++tm)
#pragma unroll
                for (int tn = 0; tn < NT; ++tn)
                    acc[tm][tn] = __builtin_amdgcn_wmma_f32_16x16x32_f16(
                        false, af[tm].v, false, bf[tn].v,
                        (short)0, acc[tm][tn], false, false);
        }
        __syncthreads();
    }

    // Epilogue: C frag -> lane hi picks M 0-7/8-15 within tile, N = r16.
#pragma unroll
    for (int tm = 0; tm < 2; ++tm) {
#pragma unroll
        for (int tn = 0; tn < NT; ++tn) {
            const int c  = n0 + wn * WN + tn * 16 + r16;
            const float bv = bias ? bias[c] : 0.0f;
#pragma unroll
            for (int v = 0; v < 8; ++v) {
                const int r = m0 + wm * 32 + tm * 16 + hi * 8 + v;
                float val = acc[tm][tn][v] * alpha + bv;
                if (act) val = fmaxf(val, 0.0f);
                const long long off = (long long)r * ldc + c;
                if (C32) C32[off] = val;
                if (C16) C16[off] = (_Float16)val;
            }
        }
    }
}

// ---------------------------------------------------------------------------
// S[g][r][n] += log(max(relu(pe[r,n,:]·Wp[g] + bp[g]), 1e-6))
// ---------------------------------------------------------------------------
__global__ void __launch_bounds__(256)
pe_logbias_add(const float* __restrict__ pe, const float* __restrict__ Wp,
               const float* __restrict__ bp, float* __restrict__ S)
{
    __shared__ float sW[G_DIM * E_DIM];
    for (int i = threadIdx.x; i < G_DIM * E_DIM; i += blockDim.x) sW[i] = Wp[i];
    __syncthreads();

    const long long idx = (long long)blockIdx.x * blockDim.x + threadIdx.x; // r*N + n
    const int n = (int)(idx & (R_DIM - 1));
    const int r = (int)(idx >> 10);

    const float4* p4 = (const float4*)(pe + idx * E_DIM);
    float accg[G_DIM];
#pragma unroll
    for (int g = 0; g < G_DIM; ++g) accg[g] = 0.0f;

    for (int e4 = 0; e4 < E_DIM / 4; ++e4) {
        float4 p = p4[e4];
#pragma unroll
        for (int g = 0; g < G_DIM; ++g) {
            float4 w = *(const float4*)(&sW[g * E_DIM + e4 * 4]);
            accg[g] += p.x * w.x + p.y * w.y + p.z * w.z + p.w * w.w;
        }
    }
#pragma unroll
    for (int g = 0; g < G_DIM; ++g) {
        float v = fmaxf(accg[g] + bp[g], 1e-6f);   // relu folded into clamp
        S[((long long)g * R_DIM + r) * R_DIM + n] += __logf(v);
    }
}

// ---------------------------------------------------------------------------
// Row softmax over N=1024, f32 in -> f16 out. 128-thread block per row.
// ---------------------------------------------------------------------------
__global__ void __launch_bounds__(128)
softmax_row_f16(const float* __restrict__ S, _Float16* __restrict__ O)
{
    __shared__ float red[128];
    const long long base = (long long)blockIdx.x * R_DIM;
    const int t = threadIdx.x;

    float vals[8];
    float lmax = -3.4e38f;
    const float4* p4 = (const float4*)(S + base + t * 8);
#pragma unroll
    for (int i = 0; i < 2; ++i) {
        float4 v = p4[i];
        vals[i * 4 + 0] = v.x; vals[i * 4 + 1] = v.y;
        vals[i * 4 + 2] = v.z; vals[i * 4 + 3] = v.w;
        lmax = fmaxf(lmax, fmaxf(fmaxf(v.x, v.y), fmaxf(v.z, v.w)));
    }
    red[t] = lmax; __syncthreads();
    for (int s = 64; s > 0; s >>= 1) {
        if (t < s) red[t] = fmaxf(red[t], red[t + s]);
        __syncthreads();
    }
    const float m = red[0]; __syncthreads();

    float lsum = 0.0f;
#pragma unroll
    for (int i = 0; i < 8; ++i) { vals[i] = __expf(vals[i] - m); lsum += vals[i]; }
    red[t] = lsum; __syncthreads();
    for (int s = 64; s > 0; s >>= 1) {
        if (t < s) red[t] += red[t + s];
        __syncthreads();
    }
    const float inv = 1.0f / red[0];

    v4h* o4 = (v4h*)(O + base + t * 8);
#pragma unroll
    for (int i = 0; i < 2; ++i) {
        v4h h;
        h.x = (_Float16)(vals[i * 4 + 0] * inv);
        h.y = (_Float16)(vals[i * 4 + 1] * inv);
        h.z = (_Float16)(vals[i * 4 + 2] * inv);
        h.w = (_Float16)(vals[i * 4 + 3] * inv);
        o4[i] = h;
    }
}

// ---------------------------------------------------------------------------
// cat16[i] = f16([ h | a | h+a ])
// ---------------------------------------------------------------------------
__global__ void __launch_bounds__(256)
concat_h_a(const float* __restrict__ h, const float* __restrict__ a,
           _Float16* __restrict__ cat)
{
    const long long idx = (long long)blockIdx.x * blockDim.x + threadIdx.x; // R*D
    const int i = (int)(idx >> 10);
    const int c = (int)(idx & (D_DIM - 1));
    const float hv = h[idx];
    const float av = a[idx];
    const long long b = (long long)i * (3 * D_DIM);
    cat[b + c]             = (_Float16)hv;
    cat[b + D_DIM + c]     = (_Float16)av;
    cat[b + 2 * D_DIM + c] = (_Float16)(hv + av);
}

// ---------------------------------------------------------------------------
// Host-side orchestration
// ---------------------------------------------------------------------------
extern "C" void kernel_launch(void* const* d_in, const int* in_sizes, int n_in,
                              void* d_out, int out_size, void* d_ws, size_t ws_size,
                              hipStream_t stream) {
    (void)in_sizes; (void)n_in; (void)out_size; (void)ws_size;

    const float* x    = (const float*)d_in[0];
    const float* pe   = (const float*)d_in[1];
    const float* Wfc1 = (const float*)d_in[2];  const float* bfc1 = (const float*)d_in[3];
    const float* Wfc2 = (const float*)d_in[4];  const float* bfc2 = (const float*)d_in[5];
    const float* Wp1  = (const float*)d_in[6];  const float* bp1  = (const float*)d_in[7];
    const float* Wq1  = (const float*)d_in[8];  const float* bq1  = (const float*)d_in[9];
    const float* Wk1  = (const float*)d_in[10]; const float* bk1  = (const float*)d_in[11];
    const float* Wo1  = (const float*)d_in[12]; const float* bo1  = (const float*)d_in[13];
    const float* Wa1  = (const float*)d_in[14]; const float* ba1  = (const float*)d_in[15];
    const float* Wp2  = (const float*)d_in[16]; const float* bp2  = (const float*)d_in[17];
    const float* Wq2  = (const float*)d_in[18]; const float* bq2  = (const float*)d_in[19];
    const float* Wk2  = (const float*)d_in[20]; const float* bk2  = (const float*)d_in[21];
    const float* Wo2  = (const float*)d_in[22]; const float* bo2  = (const float*)d_in[23];
    const float* Wa2  = (const float*)d_in[24]; const float* ba2  = (const float*)d_in[25];
    float* out = (float*)d_out;

    // ---- workspace bump allocator ----
    char* wsp = (char*)d_ws;
    auto alloc = [&](size_t bytes) -> void* {
        void* p = (void*)wsp;
        wsp += (bytes + 255) & ~(size_t)255;
        return p;
    };
    const long long RD  = (long long)R_DIM * D_DIM;          // 1M
    const long long RF  = (long long)R_DIM * FLAT_DIM;       // 12.8M
    const long long R3D = (long long)R_DIM * 3 * D_DIM;      // 3M
    const long long GRR = (long long)G_DIM * R_DIM * R_DIM;  // 16.7M

    _Float16* x16     = (_Float16*)alloc(RF  * 2);
    _Float16* wfc1_16 = (_Float16*)alloc(RF  * 2);
    _Float16* wfc2_16 = (_Float16*)alloc(RD  * 2);
    _Float16* wq1_16  = (_Float16*)alloc(RD  * 2);
    _Float16* wk1_16  = (_Float16*)alloc(RD  * 2);
    _Float16* wo1_16  = (_Float16*)alloc(RD  * 2);
    _Float16* wq2_16  = (_Float16*)alloc(RD  * 2);
    _Float16* wk2_16  = (_Float16*)alloc(RD  * 2);
    _Float16* wo2_16  = (_Float16*)alloc(RD  * 2);
    _Float16* wa1_16  = (_Float16*)alloc(R3D * 2);
    _Float16* wa2_16  = (_Float16*)alloc(R3D * 2);
    float*    h_a     = (float*)   alloc(RD  * 4);
    _Float16* h16_a   = (_Float16*)alloc(RD  * 2);
    float*    h_b     = (float*)   alloc(RD  * 4);
    _Float16* h16_b   = (_Float16*)alloc(RD  * 2);
    _Float16* q16     = (_Float16*)alloc(RD  * 2);
    _Float16* k16     = (_Float16*)alloc(RD  * 2);
    _Float16* vt16    = (_Float16*)alloc(RD  * 2);
    float*    S       = (float*)   alloc(GRR * 4);
    _Float16* attn16  = (_Float16*)alloc(GRR * 2);
    float*    aout    = (float*)   alloc(RD  * 4);
    _Float16* cat16   = (_Float16*)alloc(R3D * 2);

    auto cvt = [&](const float* src, _Float16* dst, long long n) {
        int n4 = (int)(n / 4);
        cvt_f32_to_f16<<<dim3((n4 + 255) / 256), dim3(256), 0, stream>>>(src, dst, n4);
    };
    auto gemm = [&](const _Float16* A, const _Float16* B, const float* bias,
                    float* C32, _Float16* C16, int M, int N, int K,
                    int lda, int ldb, int ldc, float alpha, int act,
                    int Z, long long bA, long long bB, long long bC, long long bBias) {
        if ((N % 128) == 0) {
            dim3 grid(N / 128, M / BM, Z);
            gemm_f16_wmma<128><<<grid, dim3(256), 0, stream>>>(
                A, B, bias, C32, C16, M, N, K, lda, ldb, ldc, alpha, act, bA, bB, bC, bBias);
        } else {
            dim3 grid(N / 64, M / BM, Z);
            gemm_f16_wmma<64><<<grid, dim3(256), 0, stream>>>(
                A, B, bias, C32, C16, M, N, K, lda, ldb, ldc, alpha, act, bA, bB, bC, bBias);
        }
    };

    // ---- precision conversion ----
    cvt(x,    x16,     RF);
    cvt(Wfc1, wfc1_16, RF);
    cvt(Wfc2, wfc2_16, RD);
    cvt(Wq1,  wq1_16,  RD);  cvt(Wk1, wk1_16, RD);  cvt(Wo1, wo1_16, RD);
    cvt(Wq2,  wq2_16,  RD);  cvt(Wk2, wk2_16, RD);  cvt(Wo2, wo2_16, RD);
    cvt(Wa1,  wa1_16,  R3D); cvt(Wa2, wa2_16, R3D);

    const float inv_sqrt_dg = 0.125f;  // 1/sqrt(64)

    auto attention = [&](const _Float16* h16, const float* Wp, const float* bp,
                         const _Float16* wq, const float* bq,
                         const _Float16* wk, const float* bk,
                         const _Float16* wo, const float* bo) {
        gemm(h16, wq, bq, nullptr, q16, R_DIM, D_DIM, D_DIM, D_DIM, D_DIM, D_DIM,
             1.0f, 0, 1, 0, 0, 0, 0);
        gemm(h16, wk, bk, nullptr, k16, R_DIM, D_DIM, D_DIM, D_DIM, D_DIM, D_DIM,
             1.0f, 0, 1, 0, 0, 0, 0);
        // Vt = Wo @ h^T (value projection hoisted before the attention apply)
        gemm(wo, h16, nullptr, nullptr, vt16, D_DIM, R_DIM, D_DIM, D_DIM, D_DIM, R_DIM,
             1.0f, 0, 1, 0, 0, 0, 0);
        // S_g = (Q_g @ K_g^T)/8, batched over 16 groups
        gemm(q16, k16, nullptr, S, nullptr, R_DIM, R_DIM, 64, D_DIM, D_DIM, R_DIM,
             inv_sqrt_dg, 0, G_DIM, 64, 64, (long long)R_DIM * R_DIM, 0);
        pe_logbias_add<<<dim3((R_DIM * R_DIM) / 256), dim3(256), 0, stream>>>(pe, Wp, bp, S);
        softmax_row_f16<<<dim3(G_DIM * R_DIM), dim3(128), 0, stream>>>(S, attn16);
        // aout[:, g*64:(g+1)*64] = A_g @ Vt_g^T + bo_g
        gemm(attn16, vt16, bo, aout, nullptr, R_DIM, 64, R_DIM, R_DIM, R_DIM, D_DIM,
             1.0f, 0, G_DIM, (long long)R_DIM * R_DIM, 64LL * R_DIM, 64, 64);
    };

    // ---- fc1 ----
    gemm(x16, wfc1_16, bfc1, h_a, h16_a, R_DIM, D_DIM, FLAT_DIM,
         FLAT_DIM, FLAT_DIM, D_DIM, 1.0f, 0, 1, 0, 0, 0, 0);
    // ---- relation block 1 ----
    attention(h16_a, Wp1, bp1, wq1_16, bq1, wk1_16, bk1, wo1_16, bo1);
    concat_h_a<<<dim3((int)(RD / 256)), dim3(256), 0, stream>>>(h_a, aout, cat16);
    gemm(cat16, wa1_16, ba1, h_b, h16_b, R_DIM, D_DIM, 3 * D_DIM,
         3 * D_DIM, 3 * D_DIM, D_DIM, 1.0f, 1 /*relu*/, 1, 0, 0, 0, 0);
    // ---- fc2 ----
    gemm(h16_b, wfc2_16, bfc2, h_a, h16_a, R_DIM, D_DIM, D_DIM,
         D_DIM, D_DIM, D_DIM, 1.0f, 0, 1, 0, 0, 0, 0);
    // ---- relation block 2 ----
    attention(h16_a, Wp2, bp2, wq2_16, bq2, wk2_16, bk2, wo2_16, bo2);
    concat_h_a<<<dim3((int)(RD / 256)), dim3(256), 0, stream>>>(h_a, aout, cat16);
    gemm(cat16, wa2_16, ba2, out, nullptr, R_DIM, D_DIM, 3 * D_DIM,
         3 * D_DIM, 3 * D_DIM, D_DIM, 1.0f, 1 /*relu*/, 1, 0, 0, 0, 0);
}
